// Block_69406671503545
// MI455X (gfx1250) — matrix operand order repaired and checked by reading, not compiled, not run on hardware
//
#include <hip/hip_runtime.h>
#include <hip/hip_bf16.h>
#include <math.h>

typedef __attribute__((ext_vector_type(16))) _Float16 v16h;
typedef __attribute__((ext_vector_type(8)))  float    v8f;

#define BB     16
#define CC     512
#define NN     625
#define NP     640
#define NHEADS 8
#define HD     64
#define HIDD   2048
#define SCL    0.125f   // hd^-0.5

// ---------------- WMMA fragment helpers (CDNA5 16x16x32 f16 layouts) ----------------

// A fragment: 16x32 f16 tile from row-major matrix (lda halfs per row).
// lane holds row = row0 + lane%16, K = k0 + 16*(lane/16) + i.
__device__ __forceinline__ v16h frag_a_rm(const _Float16* A, int lda, int row0, int k0) {
  const int lane = threadIdx.x & 31;
  const _Float16* p = A + (size_t)(row0 + (lane & 15)) * lda + (k0 + ((lane >> 4) << 4));
  v16h f;
#pragma unroll
  for (int i = 0; i < 16; ++i) f[i] = p[i];
  return f;
}

// B fragment: 32x16 f16 tile, source row-major K x N (ldb halfs per row).
// lane holds col = col0 + lane%16, K = k0 + 16*(lane/16) + i.
__device__ __forceinline__ v16h frag_b_rm(const _Float16* Bm, int ldb, int k0, int col0) {
  const int lane = threadIdx.x & 31;
  const int c  = col0 + (lane & 15);
  const int kb = k0 + ((lane >> 4) << 4);
  v16h f;
#pragma unroll
  for (int i = 0; i < 16; ++i) f[i] = Bm[(size_t)(kb + i) * ldb + c];
  return f;
}

// B fragment where B[k][col] = src[col][k] (src row-major N x K) -> contiguous loads.
__device__ __forceinline__ v16h frag_b_tr(const _Float16* src, int lds, int k0, int col0) {
  const int lane = threadIdx.x & 31;
  const _Float16* p = src + (size_t)(col0 + (lane & 15)) * lds + (k0 + ((lane >> 4) << 4));
  v16h f;
#pragma unroll
  for (int i = 0; i < 16; ++i) f[i] = p[i];
  return f;
}

__device__ __forceinline__ v8f wmma16(v16h a, v16h b, v8f c) {
  return __builtin_amdgcn_wmma_f32_16x16x32_f16(false, a, false, b, (short)0, c, false, false);
}

// ---------------- CDNA5 async global->LDS staging (ASYNCcnt path) ----------------
// ISA: generic LDS address low 32 bits are the LDS byte offset (addr[31:0]).
__device__ __forceinline__ void async_g2l_b128(void* lds_ptr, const void* gptr) {
  uint32_t lo = (uint32_t)(uintptr_t)lds_ptr;
  asm volatile("global_load_async_to_lds_b128 %0, %1, off"
               :: "v"(lo), "v"(gptr)
               : "memory");
}
__device__ __forceinline__ void wait_async0() {
  asm volatile("s_wait_asynccnt 0x0" ::: "memory");
}

// Stage a 32x64 f16 tile (rows kb..kb+31, cols c0..c0+63) of row-major B (ldb halfs)
// into LDS, fully coalesced: 256 b128 transfers shared by 128 threads.
__device__ __forceinline__ void stage_b_tile(_Float16* tile, const _Float16* Bm,
                                             int ldb, int kb, int c0) {
  const int t = threadIdx.x;   // blockDim.x == 128
#pragma unroll
  for (int h = 0; h < 2; ++h) {
    int id  = t + h * 128;     // 0..255
    int row = id >> 3;         // 0..31
    int seg = id & 7;          // 8 halfs per segment
    async_g2l_b128(tile + row * 64 + seg * 8,
                   Bm + (size_t)(kb + row) * ldb + c0 + seg * 8);
  }
  wait_async0();
}

// ---------------- elementwise / LN / conv kernels ----------------

__global__ void k_f32_to_f16(const float* __restrict__ s, _Float16* __restrict__ d, int n) {
  int i = blockIdx.x * blockDim.x + threadIdx.x;
  if (i < n) d[i] = (_Float16)s[i];
}

// Channel LayerNorm (over C at each (b,n)). torch semantics: (x-mean)/(sqrt(var)+eps)*g+b
__global__ void k_ln_f32(const float* __restrict__ x, const float* __restrict__ g,
                         const float* __restrict__ bt, float* __restrict__ out) {
  int n = blockIdx.x * blockDim.x + threadIdx.x;
  int b = blockIdx.y;
  if (n >= NN) return;
  const float* xp = x + (size_t)b * CC * NN + n;
  float s = 0.f, s2 = 0.f;
  for (int c = 0; c < CC; ++c) { float v = xp[(size_t)c * NN]; s += v; s2 += v * v; }
  float mean = s * (1.f / CC);
  float var  = fmaxf(s2 * (1.f / CC) - mean * mean, 0.f);
  float inv  = 1.f / (sqrtf(var) + 1e-5f);
  float* op = out + (size_t)b * CC * NN + n;
  for (int c = 0; c < CC; ++c)
    op[(size_t)c * NN] = (xp[(size_t)c * NN] - mean) * inv * g[c] + bt[c];
}

// LN -> f16 output padded to NP columns (zeros for n>=NN)
__global__ void k_ln_f16pad(const float* __restrict__ x, const float* __restrict__ g,
                            const float* __restrict__ bt, _Float16* __restrict__ out) {
  int n = blockIdx.x * blockDim.x + threadIdx.x;
  int b = blockIdx.y;
  if (n >= NP) return;
  _Float16* op = out + (size_t)b * CC * NP + n;
  if (n >= NN) { for (int c = 0; c < CC; ++c) op[(size_t)c * NP] = (_Float16)0.f; return; }
  const float* xp = x + (size_t)b * CC * NN + n;
  float s = 0.f, s2 = 0.f;
  for (int c = 0; c < CC; ++c) { float v = xp[(size_t)c * NN]; s += v; s2 += v * v; }
  float mean = s * (1.f / CC);
  float var  = fmaxf(s2 * (1.f / CC) - mean * mean, 0.f);
  float inv  = 1.f / (sqrtf(var) + 1e-5f);
  for (int c = 0; c < CC; ++c)
    op[(size_t)c * NP] = (_Float16)((xp[(size_t)c * NN] - mean) * inv * g[c] + bt[c]);
}

__device__ __forceinline__ void taps3x3(const float* ip, int y, int x, int Hh, int Ww, float t[9]) {
#pragma unroll
  for (int dy = -1; dy <= 1; ++dy)
#pragma unroll
    for (int dx = -1; dx <= 1; ++dx) {
      int yy = y + dy, xx = x + dx;
      t[(dy + 1) * 3 + (dx + 1)] =
          (yy >= 0 && yy < Hh && xx >= 0 && xx < Ww) ? ip[yy * Ww + xx] : 0.f;
    }
}

// dwconv3x3 on 25x25, outputs q/k/v f16 in (b, head, n_pad, hd) layout (pad rows zeroed)
__global__ void k_dwconv_qkv(const float* __restrict__ ln,
                             const float* __restrict__ wq, const float* __restrict__ bq,
                             const float* __restrict__ wk, const float* __restrict__ bk,
                             const float* __restrict__ wv, const float* __restrict__ bv,
                             _Float16* __restrict__ q, _Float16* __restrict__ k,
                             _Float16* __restrict__ v) {
  int idx = blockIdx.x * blockDim.x + threadIdx.x;
  if (idx >= BB * CC * NP) return;
  int n = idx % NP;
  int c = (idx / NP) % CC;
  int b = idx / (NP * CC);
  int head = c >> 6, d = c & 63;
  size_t o = ((size_t)(b * NHEADS + head) * NP + n) * HD + d;
  if (n >= NN) { q[o] = (_Float16)0.f; k[o] = (_Float16)0.f; v[o] = (_Float16)0.f; return; }
  const float* ip = ln + ((size_t)b * CC + c) * NN;
  float t[9];
  taps3x3(ip, n / 25, n % 25, 25, 25, t);
  const float* aq = wq + c * 9; const float* ak = wk + c * 9; const float* av = wv + c * 9;
  float sq = bq[c], sk = bk[c], sv = bv[c];
#pragma unroll
  for (int i = 0; i < 9; ++i) { sq += t[i] * aq[i]; sk += t[i] * ak[i]; sv += t[i] * av[i]; }
  q[o] = (_Float16)sq; k[o] = (_Float16)sk; v[o] = (_Float16)sv;
}

// dwconv3x3 on 5x125, outputs spe_q/k/v f16 in (b, c, n_pad) layout (pad cols zeroed)
__global__ void k_dwconv_spe(const float* __restrict__ in,
                             const float* __restrict__ wq, const float* __restrict__ bq,
                             const float* __restrict__ wk, const float* __restrict__ bk,
                             const float* __restrict__ wv, const float* __restrict__ bv,
                             _Float16* __restrict__ sq, _Float16* __restrict__ sk,
                             _Float16* __restrict__ sv) {
  int idx = blockIdx.x * blockDim.x + threadIdx.x;
  if (idx >= BB * CC * NP) return;
  int n = idx % NP;
  int c = (idx / NP) % CC;
  int b = idx / (NP * CC);
  size_t o = ((size_t)b * CC + c) * NP + n;
  if (n >= NN) { sq[o] = (_Float16)0.f; sk[o] = (_Float16)0.f; sv[o] = (_Float16)0.f; return; }
  const float* ip = in + ((size_t)b * CC + c) * NN;
  float t[9];
  taps3x3(ip, n / 125, n % 125, 5, 125, t);
  const float* aq = wq + c * 9; const float* ak = wk + c * 9; const float* av = wv + c * 9;
  float vq = bq[c], vk = bk[c], vv = bv[c];
#pragma unroll
  for (int i = 0; i < 9; ++i) { vq += t[i] * aq[i]; vk += t[i] * ak[i]; vv += t[i] * av[i]; }
  sq[o] = (_Float16)vq; sk[o] = (_Float16)vk; sv[o] = (_Float16)vv;
}

// ---------------- spatial multi-head attention (flash-style, WMMA) ----------------
// grid (40 row-tiles, 8 heads, 16 batch), block 128 (4 waves)
__global__ void k_attn(const _Float16* __restrict__ q, const _Float16* __restrict__ k,
                       const _Float16* __restrict__ v, float* __restrict__ out1) {
  __shared__ float    s_dots[16 * NP];   // 40 KB (reused as V staging in stage 3)
  __shared__ _Float16 s_attn[16 * NP];   // 20 KB
  const int mt = blockIdx.x, h = blockIdx.y, b = blockIdx.z;
  const int wave = threadIdx.x >> 5, lane = threadIdx.x & 31;
  const size_t base = (size_t)(b * NHEADS + h) * NP * HD;
  const _Float16* Q = q + base;
  const _Float16* K = k + base;
  const _Float16* V = v + base;
  const int row0 = mt * 16;

  // Stage 1: dots = scale * Q(16x64) @ K^T -> LDS, 16x640
  v16h a0 = frag_a_rm(Q, HD, row0, 0);
  v16h a1 = frag_a_rm(Q, HD, row0, 32);
  for (int jt = wave; jt < 40; jt += 4) {
    v8f acc = {};
    acc = wmma16(a0, frag_b_tr(K, HD, 0,  jt * 16), acc);
    acc = wmma16(a1, frag_b_tr(K, HD, 32, jt * 16), acc);
    int cc = jt * 16 + (lane & 15);
    int m0 = (lane >> 4) << 3;
#pragma unroll
    for (int r = 0; r < 8; ++r) s_dots[(m0 + r) * NP + cc] = acc[r] * SCL;
  }
  __syncthreads();

  // Stage 2: row softmax over 625 cols (8 threads per row), attn -> f16 with zero pad
  {
    int row = threadIdx.x >> 3, t8 = threadIdx.x & 7;
    float* rp = s_dots + row * NP;
    float mx = -1e30f;
    for (int j = t8; j < NN; j += 8) mx = fmaxf(mx, rp[j]);
#pragma unroll
    for (int off = 1; off < 8; off <<= 1) mx = fmaxf(mx, __shfl_xor(mx, off, 8));
    float sum = 0.f;
    for (int j = t8; j < NN; j += 8) { float e = __expf(rp[j] - mx); rp[j] = e; sum += e; }
#pragma unroll
    for (int off = 1; off < 8; off <<= 1) sum += __shfl_xor(sum, off, 8);
    float inv = 1.f / sum;
    _Float16* ar = s_attn + row * NP;
    for (int j = t8; j < NP; j += 8)
      ar[j] = (j < NN) ? (_Float16)(rp[j] * inv) : (_Float16)0.f;
  }
  __syncthreads();

  // Stage 3: out(16x64) = attn(16x640,f16 LDS) @ V(640x64).
  // V tiles staged via async global->LDS into the dead s_dots region.
  _Float16* vt = (_Float16*)s_dots;       // 32x64 f16 tile, 4 KB
  v8f acc = {};
  const _Float16* sa = (const _Float16*)s_attn;
  for (int kt = 0; kt < 20; ++kt) {
    stage_b_tile(vt, V, HD, kt * 32, 0);
    __syncthreads();
    acc = wmma16(frag_a_rm(sa, NP, 0, kt * 32), frag_b_rm(vt, 64, 0, wave * 16), acc);
    __syncthreads();
  }
  int d  = wave * 16 + (lane & 15);
  int m0 = row0 + ((lane >> 4) << 3);
  float* O = out1 + ((size_t)b * CC + h * HD + d) * NN;
#pragma unroll
  for (int r = 0; r < 8; ++r) { int n = m0 + r; if (n < NN) O[n] = acc[r]; }
}

// ---------------- channel attention GEMMs ----------------

// dots2[b,c,d] = scale * spe_q(512x640) @ spe_k^T ; grid (8,32,B), 4 waves = 4 n-tiles
// (both operands load contiguously: no staging needed)
__global__ void k_gemm_ca(const _Float16* __restrict__ sq, const _Float16* __restrict__ sk,
                          float* __restrict__ dots2) {
  const int wave = threadIdx.x >> 5, lane = threadIdx.x & 31;
  const int nt = blockIdx.x * 4 + wave, mt = blockIdx.y, b = blockIdx.z;
  const _Float16* A  = sq + (size_t)b * CC * NP;
  const _Float16* Bt = sk + (size_t)b * CC * NP;
  v8f acc = {};
  for (int kt = 0; kt < 20; ++kt)
    acc = wmma16(frag_a_rm(A, NP, mt * 16, kt * 32), frag_b_tr(Bt, NP, kt * 32, nt * 16), acc);
  int col = nt * 16 + (lane & 15);
  int m0  = mt * 16 + ((lane >> 4) << 3);
  float* D = dots2 + (size_t)b * CC * CC;
#pragma unroll
  for (int r = 0; r < 8; ++r) D[(size_t)(m0 + r) * CC + col] = acc[r] * SCL;
}

// row softmax over 512, one wave per row -> f16
__global__ void k_softmax512(const float* __restrict__ dots2, _Float16* __restrict__ attn2) {
  const int wave = threadIdx.x >> 5, lane = threadIdx.x & 31;
  const int rid = blockIdx.x * 4 + wave;   // 0 .. B*512-1
  const float* rp = dots2 + (size_t)rid * CC;
  float mx = -1e30f;
  float e[16];
#pragma unroll
  for (int i = 0; i < 16; ++i) mx = fmaxf(mx, rp[lane + i * 32]);
#pragma unroll
  for (int off = 1; off < 32; off <<= 1) mx = fmaxf(mx, __shfl_xor(mx, off, 32));
  float sum = 0.f;
#pragma unroll
  for (int i = 0; i < 16; ++i) { e[i] = __expf(rp[lane + i * 32] - mx); sum += e[i]; }
#pragma unroll
  for (int off = 1; off < 32; off <<= 1) sum += __shfl_xor(sum, off, 32);
  float inv = 1.f / sum;
  _Float16* op = attn2 + (size_t)rid * CC;
#pragma unroll
  for (int i = 0; i < 16; ++i) op[lane + i * 32] = (_Float16)(e[i] * inv);
}

// x_out = attn2(512x512) @ spe_v(512x640) -> f16 ; grid (10,32,B); B staged via async LDS
__global__ void k_gemm_av2(const _Float16* __restrict__ attn2, const _Float16* __restrict__ sv,
                           _Float16* __restrict__ xout) {
  __shared__ _Float16 btile[32 * 64];
  const int wave = threadIdx.x >> 5, lane = threadIdx.x & 31;
  const int mt = blockIdx.y, b = blockIdx.z;
  const int c0 = blockIdx.x * 64;
  const _Float16* A  = attn2 + (size_t)b * CC * CC;
  const _Float16* Bm = sv + (size_t)b * CC * NP;
  v8f acc = {};
  for (int kt = 0; kt < 16; ++kt) {
    stage_b_tile(btile, Bm, NP, kt * 32, c0);
    __syncthreads();
    acc = wmma16(frag_a_rm(A, CC, mt * 16, kt * 32), frag_b_rm(btile, 64, 0, wave * 16), acc);
    __syncthreads();
  }
  int col = c0 + wave * 16 + (lane & 15);
  int m0  = mt * 16 + ((lane >> 4) << 3);
  _Float16* D = xout + (size_t)b * CC * NP;
#pragma unroll
  for (int r = 0; r < 8; ++r) D[(size_t)(m0 + r) * NP + col] = (_Float16)acc[r];
}

// proj = W(512x512) @ x_out, then BN (eval) + residual -> y1 f32 ; grid (10,32,B)
__global__ void k_proj_bn_res(const _Float16* __restrict__ wp, const _Float16* __restrict__ xout,
                              const float* __restrict__ x,
                              const float* __restrict__ bng, const float* __restrict__ bnb,
                              const float* __restrict__ bnm, const float* __restrict__ bnv,
                              float* __restrict__ y1) {
  __shared__ _Float16 btile[32 * 64];
  const int wave = threadIdx.x >> 5, lane = threadIdx.x & 31;
  const int mt = blockIdx.y, b = blockIdx.z;
  const int c0 = blockIdx.x * 64;
  const _Float16* Bm = xout + (size_t)b * CC * NP;
  v8f acc = {};
  for (int kt = 0; kt < 16; ++kt) {
    stage_b_tile(btile, Bm, NP, kt * 32, c0);
    __syncthreads();
    acc = wmma16(frag_a_rm(wp, CC, mt * 16, kt * 32), frag_b_rm(btile, 64, 0, wave * 16), acc);
    __syncthreads();
  }
  int col = c0 + wave * 16 + (lane & 15);
  if (col >= NN) return;
  int m0 = mt * 16 + ((lane >> 4) << 3);
#pragma unroll
  for (int r = 0; r < 8; ++r) {
    int o = m0 + r;
    float val = (acc[r] - bnm[o]) * rsqrtf(bnv[o] + 1e-5f) * bng[o] + bnb[o];
    size_t idx = ((size_t)b * CC + o) * NN + col;
    y1[idx] = x[idx] + val;
  }
}

// FFN1: h1 = gelu(w1(2048x512) @ ln2 + b1) -> f16 ; grid (10,128,B)
__global__ void k_ffn1(const _Float16* __restrict__ w1, const _Float16* __restrict__ ln2b,
                       const float* __restrict__ b1, _Float16* __restrict__ h1) {
  __shared__ _Float16 btile[32 * 64];
  const int wave = threadIdx.x >> 5, lane = threadIdx.x & 31;
  const int mt = blockIdx.y, b = blockIdx.z;
  const int c0 = blockIdx.x * 64;
  const _Float16* Bm = ln2b + (size_t)b * CC * NP;
  v8f acc = {};
  for (int kt = 0; kt < 16; ++kt) {
    __builtin_prefetch(w1 + (size_t)(mt * 16) * CC + (kt + 1) * 32, 0, 0);
    stage_b_tile(btile, Bm, NP, kt * 32, c0);
    __syncthreads();
    acc = wmma16(frag_a_rm(w1, CC, mt * 16, kt * 32), frag_b_rm(btile, 64, 0, wave * 16), acc);
    __syncthreads();
  }
  int col = c0 + wave * 16 + (lane & 15);
  int m0  = mt * 16 + ((lane >> 4) << 3);
  _Float16* D = h1 + (size_t)b * HIDD * NP;
#pragma unroll
  for (int r = 0; r < 8; ++r) {
    int o = m0 + r;
    float vx = acc[r] + b1[o];
    float g = vx * 0.5f * (1.f + erff(vx * 0.70710678f));   // exact GELU
    D[(size_t)o * NP + col] = (_Float16)g;
  }
}

// FFN2: out = y1 + w2(512x2048) @ h1 + b2 ; grid (10,32,B)
__global__ void k_ffn2(const _Float16* __restrict__ w2, const _Float16* __restrict__ h1,
                       const float* __restrict__ b2, const float* __restrict__ y1,
                       float* __restrict__ out) {
  __shared__ _Float16 btile[32 * 64];
  const int wave = threadIdx.x >> 5, lane = threadIdx.x & 31;
  const int mt = blockIdx.y, b = blockIdx.z;
  const int c0 = blockIdx.x * 64;
  const _Float16* Bm = h1 + (size_t)b * HIDD * NP;
  v8f acc = {};
  for (int kt = 0; kt < 64; ++kt) {
    __builtin_prefetch(w2 + (size_t)(mt * 16) * HIDD + (kt + 1) * 32, 0, 0);
    stage_b_tile(btile, Bm, NP, kt * 32, c0);
    __syncthreads();
    acc = wmma16(frag_a_rm(w2, HIDD, mt * 16, kt * 32), frag_b_rm(btile, 64, 0, wave * 16), acc);
    __syncthreads();
  }
  int col = c0 + wave * 16 + (lane & 15);
  if (col >= NN) return;
  int m0 = mt * 16 + ((lane >> 4) << 3);
#pragma unroll
  for (int r = 0; r < 8; ++r) {
    int o = m0 + r;
    size_t idx = ((size_t)b * CC + o) * NN + col;
    out[idx] = y1[idx] + acc[r] + b2[o];
  }
}

// ---------------- launcher ----------------
extern "C" void kernel_launch(void* const* d_in, const int* in_sizes, int n_in,
                              void* d_out, int out_size, void* d_ws, size_t ws_size,
                              hipStream_t stream) {
  const float* x      = (const float*)d_in[0];
  const float* ln1_g  = (const float*)d_in[1];
  const float* ln1_b  = (const float*)d_in[2];
  const float* wq = (const float*)d_in[3];  const float* bq = (const float*)d_in[4];
  const float* wk = (const float*)d_in[5];  const float* bk = (const float*)d_in[6];
  const float* wv = (const float*)d_in[7];  const float* bv = (const float*)d_in[8];
  const float* swq = (const float*)d_in[9];  const float* sbq = (const float*)d_in[10];
  const float* swk = (const float*)d_in[11]; const float* sbk = (const float*)d_in[12];
  const float* swv = (const float*)d_in[13]; const float* sbv = (const float*)d_in[14];
  const float* proj_w = (const float*)d_in[15];
  const float* bn_g = (const float*)d_in[16]; const float* bn_b = (const float*)d_in[17];
  const float* bn_m = (const float*)d_in[18]; const float* bn_v = (const float*)d_in[19];
  const float* ln2_g = (const float*)d_in[20]; const float* ln2_b = (const float*)d_in[21];
  const float* w1 = (const float*)d_in[22]; const float* b1 = (const float*)d_in[23];
  const float* w2 = (const float*)d_in[24]; const float* b2 = (const float*)d_in[25];

  char* wsb = (char*)d_ws;
  size_t off = 0;
  auto take = [&](size_t bytes) -> char* {
    char* p = wsb + off;
    off = (off + bytes + 255) & ~(size_t)255;
    return p;
  };
  float*    ln1   = (float*)   take((size_t)BB * CC * NN * 4);
  _Float16* qh    = (_Float16*)take((size_t)BB * NHEADS * NP * HD * 2);
  _Float16* kh    = (_Float16*)take((size_t)BB * NHEADS * NP * HD * 2);
  _Float16* vh    = (_Float16*)take((size_t)BB * NHEADS * NP * HD * 2);
  float*    out1  = (float*)   take((size_t)BB * CC * NN * 4);
  _Float16* sq    = (_Float16*)take((size_t)BB * CC * NP * 2);
  _Float16* sk    = (_Float16*)take((size_t)BB * CC * NP * 2);
  _Float16* sv    = (_Float16*)take((size_t)BB * CC * NP * 2);
  float*    dots2 = (float*)   take((size_t)BB * CC * CC * 4);
  _Float16* attn2 = (_Float16*)take((size_t)BB * CC * CC * 2);
  _Float16* xout  = (_Float16*)take((size_t)BB * CC * NP * 2);
  float*    y1    = (float*)   take((size_t)BB * CC * NN * 4);
  _Float16* ln2bf = (_Float16*)take((size_t)BB * CC * NP * 2);
  _Float16* h1    = (_Float16*)take((size_t)BB * HIDD * NP * 2);
  _Float16* wpf   = (_Float16*)take((size_t)CC * CC * 2);
  _Float16* w1f   = (_Float16*)take((size_t)HIDD * CC * 2);
  _Float16* w2f   = (_Float16*)take((size_t)CC * HIDD * 2);

  // weight conversion (deterministic each call)
  k_f32_to_f16<<<(CC * CC + 255) / 256, 256, 0, stream>>>(proj_w, wpf, CC * CC);
  k_f32_to_f16<<<(HIDD * CC + 255) / 256, 256, 0, stream>>>(w1, w1f, HIDD * CC);
  k_f32_to_f16<<<(CC * HIDD + 255) / 256, 256, 0, stream>>>(w2, w2f, CC * HIDD);

  // LN1 -> dwconv q/k/v -> spatial attention
  k_ln_f32<<<dim3(3, BB), 256, 0, stream>>>(x, ln1_g, ln1_b, ln1);
  k_dwconv_qkv<<<(BB * CC * NP) / 256, 256, 0, stream>>>(ln1, wq, bq, wk, bk, wv, bv, qh, kh, vh);
  k_attn<<<dim3(40, NHEADS, BB), 128, 0, stream>>>(qh, kh, vh, out1);

  // spe convs -> channel attention -> proj+BN+residual
  k_dwconv_spe<<<(BB * CC * NP) / 256, 256, 0, stream>>>(out1, swq, sbq, swk, sbk, swv, sbv, sq, sk, sv);
  k_gemm_ca<<<dim3(8, 32, BB), 128, 0, stream>>>(sq, sk, dots2);
  k_softmax512<<<(BB * CC) / 4, 128, 0, stream>>>(dots2, attn2);
  k_gemm_av2<<<dim3(10, 32, BB), 128, 0, stream>>>(attn2, sv, xout);
  k_proj_bn_res<<<dim3(10, 32, BB), 128, 0, stream>>>(wpf, xout, x, bn_g, bn_b, bn_m, bn_v, y1);

  // LN2 -> FFN -> residual to d_out
  k_ln_f16pad<<<dim3(3, BB), 256, 0, stream>>>(y1, ln2_g, ln2_b, ln2bf);
  k_ffn1<<<dim3(10, 128, BB), 128, 0, stream>>>(w1f, ln2bf, b1, h1);
  k_ffn2<<<dim3(10, 32, BB), 128, 0, stream>>>(w2f, h1, b2, y1, (float*)d_out);
}
